// TransformerHead_64407329570925
// MI455X (gfx1250) — matrix operand order repaired and checked by reading, not compiled, not run on hardware
//
#include <hip/hip_runtime.h>
#include <hip/hip_bf16.h>

typedef _Float16 f16;
typedef __attribute__((ext_vector_type(16))) _Float16 v16h;
typedef __attribute__((ext_vector_type(8)))  _Float16 v8h;
typedef __attribute__((ext_vector_type(8)))  float    v8f;
typedef __attribute__((ext_vector_type(4)))  float    v4f;
typedef __attribute__((ext_vector_type(4)))  unsigned int v4u;
typedef __attribute__((ext_vector_type(8)))  int      v8i;
typedef __attribute__((ext_vector_type(4)))  int      v4i;

#define BATCH  4
#define SEQ    4096
#define DMODEL 1024
#define DK     64
#define MTOT   (BATCH * SEQ)   // 16384 total rows

// concat two 8-half fragments into the 16-half WMMA operand
__device__ __forceinline__ v16h cat16(v8h lo, v8h hi) {
  return __builtin_shufflevector(lo, hi, 0, 1, 2, 3, 4, 5, 6, 7,
                                         8, 9, 10, 11, 12, 13, 14, 15);
}

__device__ __forceinline__ v8f wmma_f16(v16h a, v16h b, v8f c) {
  // D = A(16x32 f16) * B(32x16 f16) + C(16x16 f32)
  return __builtin_amdgcn_wmma_f32_16x16x32_f16(
      false, a, false, b, (short)0, c, false, false);
}

// LDS byte offset of a __shared__ object (generic LDS addr: offset in [31:0])
__device__ __forceinline__ unsigned int lds_off(const void* p) {
  return (unsigned int)(unsigned long long)p;
}

// -----------------------------------------------------------------------------
// Tensor Data Mover: 2D tile load Global -> LDS (CDNA5 D#, ISA 08 section 8).
//   data_size code: 0=1B 1=2B 2=4B 3=8B; dims/strides in data_size units.
// Issue from ONE wave only; completion via s_wait_tensorcnt.
// This toolchain exposes the 6-arg builtin: (g0, g1, g2, g3, g_extra, cpol).
// -----------------------------------------------------------------------------
__device__ __forceinline__ void tdm_load_2d(unsigned int lds_addr,
                                            const void* gptr,
                                            unsigned int dsz_code,
                                            unsigned int tile_d0,
                                            unsigned int tile_d1,
                                            unsigned int tensor_d0,
                                            unsigned int tensor_d1,
                                            unsigned long long stride0) {
  unsigned long long ga = (unsigned long long)gptr;
  v4u g0;
  g0[0] = 1u;                                            // count=1 (valid), user
  g0[1] = lds_addr;                                      // lds_addr (bytes)
  g0[2] = (unsigned int)(ga & 0xFFFFFFFFu);              // global_addr[31:0]
  g0[3] = (unsigned int)((ga >> 32) & 0x01FFFFFFu)       // global_addr[56:32]
          | (2u << 30);                                  // type=2 ("image")
  v8i g1;
  g1[0] = (int)(dsz_code << 16);                         // workgroup_mask=0,data_size
  g1[1] = (int)((tensor_d0 & 0xFFFFu) << 16);            // tensor_dim0[15:0]
  g1[2] = (int)(((tensor_d0 >> 16) & 0xFFFFu) |          // tensor_dim0[31:16]
                ((tensor_d1 & 0xFFFFu) << 16));          // tensor_dim1[15:0]
  g1[3] = (int)(((tensor_d1 >> 16) & 0xFFFFu) |          // tensor_dim1[31:16]
                ((tile_d0 & 0xFFFFu) << 16));            // tile_dim0
  g1[4] = (int)(tile_d1 & 0xFFFFu);                      // tile_dim1 (tile_dim2=0)
  g1[5] = (int)(unsigned int)(stride0 & 0xFFFFFFFFu);    // dim0_stride[31:0]
  g1[6] = (int)(unsigned int)((stride0 >> 32) & 0xFFFFu);// dim0_stride[47:32]
  g1[7] = 0;                                             // dim1_stride unused (2D)
  v4i g2 = {};                                           // 2D: dims 2..4 unused
  v4i g3 = {};
  v8i g4 = {};                                           // extra group (unused)
  __builtin_amdgcn_tensor_load_to_lds(g0, g1, g2, g3, g4, 0);
}

// -----------------------------------------------------------------------------
// Kernel 1: Q = x*Wq^T, K = x*Wk^T (f16 out, row-major [MTOT][64]).
// 8 waves: waves 0-3 -> Q n-tiles, waves 4-7 -> K n-tiles. The x tile
// (16 rows x 32 k, f32) is TDM double-buffered into LDS by wave 0.
// -----------------------------------------------------------------------------
__global__ void __launch_bounds__(256)
proj_qk_kernel(const float* __restrict__ x, const float* __restrict__ Wq,
               const float* __restrict__ Wk, f16* __restrict__ Qh,
               f16* __restrict__ Kh) {
  const int tid  = threadIdx.x;
  const int wave = tid >> 5;
  const int lane = tid & 31;
  const int half = lane >> 4;
  const int lr   = lane & 15;
  const int mb   = blockIdx.x * 16;

  const float* W   = (wave < 4) ? Wq : Wk;
  f16*         Out = (wave < 4) ? Qh : Kh;
  const int    n0  = (wave & 3) * 16;

  __shared__ __align__(16) float xbuf[2][16 * 32];   // TDM double buffer (f32)

  v8f zero = {};
  v8f acc  = zero;
  const int klo = half * 16;
  const int kb  = half * 16;

  if (wave == 0) {
    tdm_load_2d(lds_off(&xbuf[0][0]), x + (size_t)mb * DMODEL,
                2 /*4B*/, 32, 16, 32, 16, DMODEL);
    __builtin_amdgcn_s_wait_tensorcnt(0);
  }
  __syncthreads();

  for (int it = 0; it < DMODEL / 32; ++it) {
    const int buf = it & 1;
    if (wave == 0 && it + 1 < DMODEL / 32)   // overlap next tile with compute
      tdm_load_2d(lds_off(&xbuf[buf ^ 1][0]),
                  x + (size_t)mb * DMODEL + (it + 1) * 32,
                  2 /*4B*/, 32, 16, 32, 16, DMODEL);

    // A fragment: row lr, K {klo..klo+15} of this tile
    v16h a;
    {
      const v4f* lo = (const v4f*)&xbuf[buf][lr * 32 + klo];
      const v4f* hi = (const v4f*)&xbuf[buf][lr * 32 + klo + 8];
      v4f a0 = lo[0], a1 = lo[1], a2 = hi[0], a3 = hi[1];
#pragma unroll
      for (int j = 0; j < 4; ++j) {
        a[j]      = (f16)a0[j];
        a[4 + j]  = (f16)a1[j];
        a[8 + j]  = (f16)a2[j];
        a[12 + j] = (f16)a3[j];
      }
    }

    // B[k][n] = W[n][k]; row of W contiguous -> 4 x b128 + cvt
    const v4f* wp = (const v4f*)(W + (size_t)(n0 + lr) * DMODEL + it * 32 + kb);
    v16h b;
#pragma unroll
    for (int i = 0; i < 4; ++i) {
      v4f t = wp[i];
#pragma unroll
      for (int j = 0; j < 4; ++j) b[i * 4 + j] = (f16)t[j];
    }

    acc = wmma_f16(a, b, acc);

    if (wave == 0) __builtin_amdgcn_s_wait_tensorcnt(0);
    __syncthreads();
  }

#pragma unroll
  for (int r = 0; r < 8; ++r)
    Out[(size_t)(mb + half * 8 + r) * DK + n0 + lr] = (f16)acc[r];
}

// -----------------------------------------------------------------------------
// Kernel 2: V = x*Wv^T stored TRANSPOSED Vt[n][m] (f16 [DMODEL][MTOT]).
// Same TDM-staged x tile; wave w covers cols [128w,128w+128) = 8 tiles.
// -----------------------------------------------------------------------------
__global__ void __launch_bounds__(256)
proj_v_kernel(const float* __restrict__ x, const float* __restrict__ Wv,
              f16* __restrict__ Vt) {
  const int tid  = threadIdx.x;
  const int wave = tid >> 5;
  const int lane = tid & 31;
  const int half = lane >> 4;
  const int lr   = lane & 15;
  const int mb   = blockIdx.x * 16;

  __shared__ __align__(16) float xbuf[2][16 * 32];

  v8f zero = {};
  v8f acc[8];
#pragma unroll
  for (int t = 0; t < 8; ++t) acc[t] = zero;

  const int klo = half * 16;
  const int kb  = half * 16;

  if (wave == 0) {
    tdm_load_2d(lds_off(&xbuf[0][0]), x + (size_t)mb * DMODEL,
                2, 32, 16, 32, 16, DMODEL);
    __builtin_amdgcn_s_wait_tensorcnt(0);
  }
  __syncthreads();

  for (int it = 0; it < DMODEL / 32; ++it) {
    const int buf = it & 1;
    if (wave == 0 && it + 1 < DMODEL / 32)
      tdm_load_2d(lds_off(&xbuf[buf ^ 1][0]),
                  x + (size_t)mb * DMODEL + (it + 1) * 32,
                  2, 32, 16, 32, 16, DMODEL);

    v16h a;
    {
      const v4f* lo = (const v4f*)&xbuf[buf][lr * 32 + klo];
      const v4f* hi = (const v4f*)&xbuf[buf][lr * 32 + klo + 8];
      v4f a0 = lo[0], a1 = lo[1], a2 = hi[0], a3 = hi[1];
#pragma unroll
      for (int j = 0; j < 4; ++j) {
        a[j]      = (f16)a0[j];
        a[4 + j]  = (f16)a1[j];
        a[8 + j]  = (f16)a2[j];
        a[12 + j] = (f16)a3[j];
      }
    }

#pragma unroll
    for (int t = 0; t < 8; ++t) {
      const int n0 = wave * 128 + t * 16;
      const v4f* wp =
          (const v4f*)(Wv + (size_t)(n0 + lr) * DMODEL + it * 32 + kb);
      v16h b;
#pragma unroll
      for (int i = 0; i < 4; ++i) {
        v4f tt = wp[i];
#pragma unroll
        for (int j = 0; j < 4; ++j) b[i * 4 + j] = (f16)tt[j];
      }
      acc[t] = wmma_f16(a, b, acc[t]);
    }

    if (wave == 0) __builtin_amdgcn_s_wait_tensorcnt(0);
    __syncthreads();
  }

#pragma unroll
  for (int t = 0; t < 8; ++t)
#pragma unroll
    for (int r = 0; r < 8; ++r)
      Vt[(size_t)(wave * 128 + t * 16 + lr) * MTOT + mb + half * 8 + r] =
          (f16)acc[t][r];
}

// -----------------------------------------------------------------------------
// Kernel 3: flash attention, 32 query rows per block, 64 keys per iteration.
// K tiles (64x64 f16 = 8KB) are TDM double-buffered into LDS; the next tile's
// DMA overlaps the softmax of the current one. V read directly from the
// transposed Vt (contiguous b128, L2-resident).
// -----------------------------------------------------------------------------
__global__ void __launch_bounds__(256)
attn_kernel(const f16* __restrict__ Qh, const f16* __restrict__ Kh,
            const f16* __restrict__ Vt, float* __restrict__ out) {
  const int tid  = threadIdx.x;
  const int wave = tid >> 5;
  const int lane = tid & 31;
  const int half = lane >> 4;
  const int lr   = lane & 15;
  const int mb   = blockIdx.x * 32;
  const int kb0  = (mb >> 12) << 12;    // batch key base (S = 4096)

  __shared__ __align__(16) f16 kbuf[2][64 * DK];   // TDM K double buffer (16KB)
  __shared__ float s_lds[32][65];
  __shared__ __align__(16) f16 p_lds[32][64];
  __shared__ float m_lds[32], l_lds[32], alpha_lds[32];

  const int smi = wave >> 2;
  const int ski = wave & 3;

  // Q fragments for this wave's score tile (loop-invariant)
  v16h qsf[2];
#pragma unroll
  for (int ks = 0; ks < 2; ++ks) {
    const f16* qp = Qh + (size_t)(mb + smi * 16 + lr) * DK + ks * 32 + half * 8;
    qsf[ks] = cat16(*(const v8h*)qp, *(const v8h*)(qp + 16));
  }

  v8f zero = {};
  v8f acc[2][8];
#pragma unroll
  for (int mi = 0; mi < 2; ++mi)
#pragma unroll
    for (int t = 0; t < 8; ++t) acc[mi][t] = zero;

  if (wave == 0) {
    tdm_load_2d(lds_off(&kbuf[0][0]), Kh + (size_t)kb0 * DK,
                1 /*2B*/, DK, 64, DK, 64, DK);
    if (tid < 32) { m_lds[tid] = -3.0e38f; l_lds[tid] = 0.f; }
    __builtin_amdgcn_s_wait_tensorcnt(0);
  }
  __syncthreads();

  for (int jb = 0; jb < SEQ; jb += 64) {
    const int buf = (jb >> 6) & 1;

    // ---- scores from the LDS K tile (each wave one 16x16 tile) ----
    v8f s = zero;
#pragma unroll
    for (int ks = 0; ks < 2; ++ks) {
      const f16* kp = &kbuf[buf][(ski * 16 + lr) * DK + ks * 32 + half * 8];
      v16h kf = cat16(*(const v8h*)kp, *(const v8h*)(kp + 16));
      s = wmma_f16(qsf[ks], kf, s);
    }
#pragma unroll
    for (int r = 0; r < 8; ++r)
      s_lds[smi * 16 + half * 8 + r][ski * 16 + lr] = s[r];

    if (jb + 64 < SEQ)   // warm L2 for the next V tile
      __builtin_prefetch(Vt + (size_t)(wave * 128 + lr) * MTOT + kb0 + jb + 64,
                         0, 0);
    __syncthreads();   // B1: s_lds ready; kbuf[buf^1] no longer read by anyone

    // ---- wave 0: kick next K DMA, run online softmax, then wait DMA ----
    if (wave == 0) {
      if (jb + 64 < SEQ)
        tdm_load_2d(lds_off(&kbuf[buf ^ 1][0]),
                    Kh + (size_t)(kb0 + jb + 64) * DK,
                    1, DK, 64, DK, 64, DK);
      float mold = m_lds[tid];
      float mnew = mold;
#pragma unroll 8
      for (int c = 0; c < 64; ++c)
        mnew = fmaxf(mnew, s_lds[tid][c] * 0.125f);
      float alpha = __expf(mold - mnew);
      float lsum  = 0.f;
#pragma unroll 8
      for (int c = 0; c < 64; ++c) {
        float p = __expf(s_lds[tid][c] * 0.125f - mnew);
        lsum += p;
        p_lds[tid][c] = (f16)p;
      }
      m_lds[tid]     = mnew;
      l_lds[tid]     = l_lds[tid] * alpha + lsum;
      alpha_lds[tid] = alpha;
      __builtin_amdgcn_s_wait_tensorcnt(0);
    }
    __syncthreads();   // B2: p/alpha ready AND next K tile resident

    // ---- rescale accumulators ----
#pragma unroll
    for (int mi = 0; mi < 2; ++mi) {
      float al[8];
#pragma unroll
      for (int r = 0; r < 8; ++r) al[r] = alpha_lds[mi * 16 + half * 8 + r];
#pragma unroll
      for (int t = 0; t < 8; ++t)
#pragma unroll
        for (int r = 0; r < 8; ++r) acc[mi][t][r] *= al[r];
    }

    // ---- O += P * V ----
#pragma unroll
    for (int ks = 0; ks < 2; ++ks) {
      v16h pf[2];
#pragma unroll
      for (int mi = 0; mi < 2; ++mi) {
        const f16* pp = &p_lds[mi * 16 + lr][ks * 32 + half * 8];
        pf[mi] = cat16(*(const v8h*)pp, *(const v8h*)(pp + 16));
      }
#pragma unroll
      for (int t = 0; t < 8; ++t) {
        const f16* vp = Vt + (size_t)(wave * 128 + t * 16 + lr) * MTOT + kb0 +
                        jb + ks * 32 + half * 8;
        v16h vf = cat16(*(const v8h*)vp, *(const v8h*)(vp + 16));
#pragma unroll
        for (int mi = 0; mi < 2; ++mi)
          acc[mi][t] = wmma_f16(pf[mi], vf, acc[mi][t]);
      }
    }
  }

  __syncthreads();
#pragma unroll
  for (int mi = 0; mi < 2; ++mi) {
#pragma unroll
    for (int r = 0; r < 8; ++r) {
      const int   row = mb + mi * 16 + half * 8 + r;
      const float inv = 1.0f / l_lds[mi * 16 + half * 8 + r];
#pragma unroll
      for (int t = 0; t < 8; ++t)
        out[(size_t)row * DMODEL + wave * 128 + t * 16 + lr] =
            acc[mi][t][r] * inv;
    }
  }
}

// -----------------------------------------------------------------------------
// Launch: Qh (2MB) | Kh (2MB) | Vt (32MB) in workspace (needs >= 36MB).
// -----------------------------------------------------------------------------
extern "C" void kernel_launch(void* const* d_in, const int* in_sizes, int n_in,
                              void* d_out, int out_size, void* d_ws,
                              size_t ws_size, hipStream_t stream) {
  (void)in_sizes; (void)n_in; (void)out_size; (void)ws_size;
  const float* x  = (const float*)d_in[0];
  const float* Wq = (const float*)d_in[1];
  const float* Wk = (const float*)d_in[2];
  const float* Wv = (const float*)d_in[3];
  float* out = (float*)d_out;

  f16* Qh = (f16*)d_ws;
  f16* Kh = Qh + (size_t)MTOT * DK;
  f16* Vt = Kh + (size_t)MTOT * DK;

  proj_qk_kernel<<<dim3(MTOT / 16), dim3(256), 0, stream>>>(x, Wq, Wk, Qh, Kh);
  proj_v_kernel<<<dim3(MTOT / 16), dim3(256), 0, stream>>>(x, Wv, Vt);
  attn_kernel<<<dim3(MTOT / 32), dim3(256), 0, stream>>>(Qh, Kh, Vt, out);
}
